// EpisodicModule_20289425506936
// MI455X (gfx1250) — matrix operand order repaired and checked by reading, not compiled, not run on hardware
//
#include <hip/hip_runtime.h>
#include <math.h>

typedef __attribute__((ext_vector_type(16))) __bf16 v16bf;
typedef __attribute__((ext_vector_type(8)))  float  v8f;

#define T_DIM 2048
#define BATCH 10
#define WD    300
#define ATT_H 16
#define NH    900    // 3*HID
#define GH_PITCH 912
#define KC2   29     // gate K chunks: 900 -> 928
#define KC45  10     // K=300 -> 320
#define NT45  57     // N tiles for 900
#define TT2   128    // T_DIM/16

// 16-bit A-matrix 16x32 layout (ISA 7.12.2): element e of v16bf -> K offset
__device__ __forceinline__ int amap_k(int e, int hi) {
  return (e < 8) ? (hi * 8 + e) : (hi * 8 + 8 + e);
}
// inverse: K offset (0..31) -> (hi, e)
__device__ __forceinline__ void ainv_k(int off, int& hi, int& e) {
  if (off < 8)       { hi = 0; e = off; }
  else if (off < 16) { hi = 1; e = off - 8; }
  else if (off < 24) { hi = 0; e = off - 8; }
  else               { hi = 1; e = off - 16; }
}

// ---------------- Kernel 1: fold constant z-blocks into per-b weights ------
__global__ void k1_precompute(const float* __restrict__ m, const float* __restrict__ q,
                              const float* __restrict__ Wb, const float* __restrict__ W1,
                              const float* __restrict__ b1,
                              float* __restrict__ Weff, float* __restrict__ c16) {
  __shared__ float tmp1[BATCH * ATT_H];
  __shared__ float tmp2[BATCH * ATT_H];
  __shared__ float Mq[WD * ATT_H];
  __shared__ float Mm[WD * ATT_H];
  const int tid = threadIdx.x;
  for (int idx = tid; idx < BATCH * ATT_H; idx += blockDim.x) {
    int k = idx / ATT_H, h = idx % ATT_H;
    float s1 = 0.f, s2 = 0.f;
    for (int d = 0; d < WD; ++d) {
      s1 += q[k * WD + d] * W1[(7 * WD + d) * ATT_H + h];
      s2 += m[k * WD + d] * W1[(8 * WD + d) * ATT_H + h];
    }
    tmp1[idx] = s1; tmp2[idx] = s2;
  }
  __syncthreads();
  for (int idx = tid; idx < WD * ATT_H; idx += blockDim.x) {
    int d = idx / ATT_H, h = idx % ATT_H;
    float s1 = 0.f, s2 = 0.f;
    for (int k = 0; k < BATCH; ++k) {
      float w = Wb[d * BATCH + k];
      s1 += w * tmp1[k * ATT_H + h];
      s2 += w * tmp2[k * ATT_H + h];
    }
    Mq[idx] = s1; Mm[idx] = s2;
  }
  __syncthreads();
  for (int idx = tid; idx < BATCH * WD * ATT_H; idx += blockDim.x) {
    int b = idx / (WD * ATT_H);
    int rem = idx % (WD * ATT_H);
    int d = rem / ATT_H, h = rem % ATT_H;
    Weff[idx] = W1[d * ATT_H + h]
              + q[b * WD + d] * W1[(3 * WD + d) * ATT_H + h]
              + m[b * WD + d] * W1[(4 * WD + d) * ATT_H + h]
              + Mq[rem] + Mm[rem];
  }
  for (int idx = tid; idx < BATCH * ATT_H; idx += blockDim.x) {
    int b = idx / ATT_H, h = idx % ATT_H;
    float s = b1[h];
    for (int d = 0; d < WD; ++d) {
      s += m[b * WD + d] * W1[(1 * WD + d) * ATT_H + h]
         + q[b * WD + d] * W1[(2 * WD + d) * ATT_H + h];
    }
    c16[idx] = s;
  }
}

// ---------------- Pack kernels: swizzle operands into WMMA fragments -------
// Fragment layout everywhere: dst[frag][lane(32)][e(16)] bf16, dst[idx] linear.

// gate A: [cs | |cs-q| | |cs-m|] per b, rows=t, K=900->928
__global__ void k_pack_gateA(const float* __restrict__ cs, const float* __restrict__ m,
                             const float* __restrict__ q, __bf16* __restrict__ dst) {
  const int TOT = BATCH * TT2 * KC2 * 512;
  int idx = blockIdx.x * blockDim.x + threadIdx.x;
  if (idx >= TOT) return;
  int e = idx & 15, lane = (idx >> 4) & 31;
  int frag = idx >> 9;
  int kc = frag % KC2, tt = (frag / KC2) % TT2, b = frag / (KC2 * TT2);
  int nloc = lane & 15, hi = lane >> 4;
  int t = tt * 16 + nloc;
  int k = kc * 32 + amap_k(e, hi);
  float v = 0.f;
  if (k < WD)          v = cs[t * (BATCH * WD) + b * WD + k];
  else if (k < 2 * WD) { int d = k - WD;     v = fabsf(cs[t * (BATCH * WD) + b * WD + d] - q[b * WD + d]); }
  else if (k < 3 * WD) { int d = k - 2 * WD; v = fabsf(cs[t * (BATCH * WD) + b * WD + d] - m[b * WD + d]); }
  dst[idx] = (__bf16)v;
}

// gate B: [Weff(b) | W1_|cs-q| | W1_|cs-m|], K=900->928, N=16
__global__ void k_pack_gateB(const float* __restrict__ Weff, const float* __restrict__ W1,
                             __bf16* __restrict__ dst) {
  const int TOT = BATCH * KC2 * 512;
  int idx = blockIdx.x * blockDim.x + threadIdx.x;
  if (idx >= TOT) return;
  int e = idx & 15, lane = (idx >> 4) & 31;
  int frag = idx >> 9;
  int kc = frag % KC2, b = frag / KC2;
  int nloc = lane & 15, hi = lane >> 4;
  int kB = kc * 32 + amap_k(e, hi);
  float v = 0.f;
  if (kB < WD)          v = Weff[(b * WD + kB) * ATT_H + nloc];
  else if (kB < 2 * WD) v = W1[(5 * WD + (kB - WD)) * ATT_H + nloc];
  else if (kB < 3 * WD) v = W1[(6 * WD + (kB - 2 * WD)) * ATT_H + nloc];
  dst[idx] = (__bf16)v;
}

// seq A for gi GEMM: rows = flattened (t,i), K=300->320
__global__ void k_pack_giA(const float* __restrict__ cs, const float* __restrict__ Gsoft,
                           __bf16* __restrict__ dst) {
  const int TOT = (T_DIM * BATCH / 16) * KC45 * 512;
  int idx = blockIdx.x * blockDim.x + threadIdx.x;
  if (idx >= TOT) return;
  int e = idx & 15, lane = (idx >> 4) & 31;
  int frag = idx >> 9;
  int kc = frag % KC45, rt = frag / KC45;
  int nloc = lane & 15, hi = lane >> 4;
  int row = rt * 16 + nloc;
  int t = row / BATCH, i = row % BATCH;
  int k = kc * 32 + amap_k(e, hi);
  float v = 0.f;
  if (k < WD) {
    int f = i * WD + k;
    int col = f % BATCH, dr = f / BATCH;
    v = Gsoft[t * BATCH + col] * cs[t * (BATCH * WD) + col * WD + dr];
  }
  dst[idx] = (__bf16)v;
}

// generic [300 x 900] weight -> B fragments (Wih, Whh)
__global__ void k_pack_w(const float* __restrict__ src, __bf16* __restrict__ dst) {
  const int TOT = NT45 * KC45 * 512;
  int idx = blockIdx.x * blockDim.x + threadIdx.x;
  if (idx >= TOT) return;
  int e = idx & 15, lane = (idx >> 4) & 31;
  int frag = idx >> 9;
  int kc = frag % KC45, nt = frag / KC45;
  int nloc = lane & 15, hi = lane >> 4;
  int k = kc * 32 + amap_k(e, hi);
  int n = nt * 16 + nloc;
  float v = (k < WD && n < NH) ? src[k * NH + n] : 0.f;
  dst[idx] = (__bf16)v;
}

// ---------------- Kernel 2: attention gate (pure WMMA stream) --------------
__global__ void k2_gate(const __bf16* __restrict__ A2, const __bf16* __restrict__ B2,
                        const float* __restrict__ W2, const float* __restrict__ b2,
                        const float* __restrict__ c16, const float* __restrict__ maskp,
                        float* __restrict__ Gout) {
  const int b = blockIdx.y;
  const int tt = blockIdx.x;
  const int lane = threadIdx.x & 31;
  const int nloc = lane & 15, hi = lane >> 4;
  const __bf16* ap = A2 + ((size_t)(b * TT2 + tt) * KC2) * 512 + lane * 16;
  const __bf16* bp = B2 + ((size_t)b * KC2) * 512 + lane * 16;
  v8f acc = {};
  for (int kc = 0; kc < KC2; ++kc) {
    v16bf av = *reinterpret_cast<const v16bf*>(ap + kc * 512);
    v16bf bv = *reinterpret_cast<const v16bf*>(bp + kc * 512);
    acc = __builtin_amdgcn_wmma_f32_16x16x32_bf16(false, av, false, bv, (short)0, acc, false, false);
  }
  float w2h = W2[nloc];
  float c16v = c16[b * ATT_H + nloc];
  float part[8];
#pragma unroll
  for (int r = 0; r < 8; ++r) part[r] = tanhf(acc[r] + c16v) * w2h;
#pragma unroll
  for (int msk = 1; msk < 16; msk <<= 1) {
#pragma unroll
    for (int r = 0; r < 8; ++r) part[r] += __shfl_xor(part[r], msk, 32);
  }
  if (nloc == 0) {
    float bb2 = b2[0];
#pragma unroll
    for (int r = 0; r < 8; ++r) {
      int tt2 = tt * 16 + hi * 8 + r;
      float g = 1.f / (1.f + expf(-(part[r] + bb2)));
      Gout[b * T_DIM + tt2] = g * maskp[tt2 * BATCH + b];
    }
  }
}

// ---------------- Kernel 3: softmax over t per b ---------------------------
__global__ void k3_softmax(const float* __restrict__ Gout, float* __restrict__ Gsoft) {
  const int b = blockIdx.x;
  __shared__ float red[256];
  const int tid = threadIdx.x;
  float mx = -1e30f;
  for (int t = tid; t < T_DIM; t += 256) mx = fmaxf(mx, Gout[b * T_DIM + t]);
  red[tid] = mx; __syncthreads();
  for (int s = 128; s > 0; s >>= 1) { if (tid < s) red[tid] = fmaxf(red[tid], red[tid + s]); __syncthreads(); }
  mx = red[0]; __syncthreads();
  float sum = 0.f;
  for (int t = tid; t < T_DIM; t += 256) sum += expf(Gout[b * T_DIM + t] - mx);
  red[tid] = sum; __syncthreads();
  for (int s = 128; s > 0; s >>= 1) { if (tid < s) red[tid] += red[tid + s]; __syncthreads(); }
  float inv = 1.f / red[0];
  for (int t = tid; t < T_DIM; t += 256) Gsoft[t * BATCH + b] = expf(Gout[b * T_DIM + t] - mx) * inv;
}

// ---------------- Kernel 4: gi = seq @ Wih + bih (pure WMMA stream) --------
__global__ void k4_gi(const __bf16* __restrict__ A4, const __bf16* __restrict__ B4,
                      const float* __restrict__ bih, float* __restrict__ gi) {
  const int rt = blockIdx.x, nt = blockIdx.y;
  const int lane = threadIdx.x & 31;
  const int nloc = lane & 15, hi = lane >> 4;
  const __bf16* ap = A4 + ((size_t)rt * KC45) * 512 + lane * 16;
  const __bf16* bp = B4 + ((size_t)nt * KC45) * 512 + lane * 16;
  v8f acc = {};
#pragma unroll
  for (int kc = 0; kc < KC45; ++kc) {
    v16bf av = *reinterpret_cast<const v16bf*>(ap + kc * 512);
    v16bf bv = *reinterpret_cast<const v16bf*>(bp + kc * 512);
    acc = __builtin_amdgcn_wmma_f32_16x16x32_bf16(false, av, false, bv, (short)0, acc, false, false);
  }
  int n = nt * 16 + nloc;
  if (n < NH) {
    float bn = bih[n];
#pragma unroll
    for (int r = 0; r < 8; ++r) {
      int orow = rt * 16 + hi * 8 + r;
      gi[(size_t)orow * NH + n] = acc[r] + bn;
    }
  }
}

// ---------------- Kernel 5: sequential GRU, persistent workgroup -----------
__global__ void __launch_bounds__(1024) k5_gru(const float* __restrict__ m,
                      const __bf16* __restrict__ B5, const float* __restrict__ bhh,
                      const float* __restrict__ gi, float* __restrict__ hout) {
  __shared__ float hbuf[BATCH * WD];
  __shared__ float gh[BATCH * GH_PITCH];
  __shared__ __align__(32) __bf16 abuf[KC45 * 32 * 16];  // h in A-fragment layout
  const int tid = threadIdx.x;
  const int wave = tid >> 5;
  const int lane = tid & 31;
  const int nloc = lane & 15, hi = lane >> 4;
  // zero A fragments once (pad rows 10..15, K>=300 stay zero forever)
  for (int idx = tid; idx < KC45 * 32 * 16; idx += 1024) abuf[idx] = (__bf16)0.f;
  // init h = m, packed into fragments too
  for (int idx = tid; idx < BATCH * WD; idx += 1024) {
    float v = m[idx];
    hbuf[idx] = v;
    int i = idx / WD, d = idx % WD;
    int c = d >> 5, off = d & 31, ahi, ae;
    ainv_k(off, ahi, ae);
    abuf[((c * 32) + (ahi * 16 + i)) * 16 + ae] = (__bf16)v;
  }
  __syncthreads();
  for (int t = 0; t < T_DIM; ++t) {
    // gh = h @ Whh : 57 N-tiles over 32 waves, fragments streamed
    for (int nt = wave; nt < NT45; nt += 32) {
      const __bf16* bp = B5 + ((size_t)nt * KC45) * 512 + lane * 16;
      v8f acc = {};
#pragma unroll
      for (int kc = 0; kc < KC45; ++kc) {
        v16bf av = *reinterpret_cast<const v16bf*>(&abuf[(kc * 32 + lane) * 16]);
        v16bf bv = *reinterpret_cast<const v16bf*>(bp + kc * 512);
        acc = __builtin_amdgcn_wmma_f32_16x16x32_bf16(false, av, false, bv, (short)0, acc, false, false);
      }
#pragma unroll
      for (int r = 0; r < 8; ++r) {
        int mrow = hi * 8 + r;
        if (mrow < BATCH) gh[mrow * GH_PITCH + nt * 16 + nloc] = acc[r];
      }
    }
    __syncthreads();
    // elementwise GRU update; stage h_new in registers, then commit
    const float* git = gi + (size_t)t * BATCH * NH;
    float hn[3];
    int cnt = 0;
    for (int idx = tid; idx < BATCH * WD; idx += 1024) {
      int i = idx / WD, d = idx % WD;
      float gir = git[i * NH + d];
      float giz = git[i * NH + WD + d];
      float gin = git[i * NH + 2 * WD + d];
      float ghr = gh[i * GH_PITCH + d] + bhh[d];
      float ghz = gh[i * GH_PITCH + WD + d] + bhh[WD + d];
      float ghn = gh[i * GH_PITCH + 2 * WD + d] + bhh[2 * WD + d];
      float rr = 1.f / (1.f + expf(-(gir + ghr)));
      float zz = 1.f / (1.f + expf(-(giz + ghz)));
      float nn = tanhf(gin + rr * ghn);
      hn[cnt++] = (1.f - zz) * nn + zz * hbuf[idx];
    }
    __syncthreads();
    cnt = 0;
    for (int idx = tid; idx < BATCH * WD; idx += 1024) {
      float v = hn[cnt++];
      hbuf[idx] = v;
      int i = idx / WD, d = idx % WD;
      int c = d >> 5, off = d & 31, ahi, ae;
      ainv_k(off, ahi, ae);
      abuf[((c * 32) + (ahi * 16 + i)) * 16 + ae] = (__bf16)v;
    }
    __syncthreads();
  }
  for (int idx = tid; idx < BATCH * WD; idx += 1024) hout[idx] = hbuf[idx];
}

extern "C" void kernel_launch(void* const* d_in, const int* in_sizes, int n_in,
                              void* d_out, int out_size, void* d_ws, size_t ws_size,
                              hipStream_t stream) {
  const float* cs  = (const float*)d_in[0];
  const float* m   = (const float*)d_in[1];
  const float* q   = (const float*)d_in[2];
  const float* msk = (const float*)d_in[3];
  const float* Wb  = (const float*)d_in[4];
  const float* W1  = (const float*)d_in[5];
  const float* b1  = (const float*)d_in[6];
  const float* W2  = (const float*)d_in[7];
  const float* b2  = (const float*)d_in[8];
  const float* Wih = (const float*)d_in[9];
  const float* Whh = (const float*)d_in[10];
  const float* bih = (const float*)d_in[11];
  const float* bhh = (const float*)d_in[12];

  float* out  = (float*)d_out;
  float* G    = out;                     // [B, T]
  float* hfin = out + BATCH * T_DIM;     // [B, WD]

  // workspace carve-out (256B aligned)
  char* base = (char*)d_ws;
  size_t off = 0;
#define WS_ALLOC(ptr, type, count)                         \
  type* ptr = (type*)(base + off);                         \
  off = (off + (size_t)(count) * sizeof(type) + 255) & ~(size_t)255;

  WS_ALLOC(Weff,  float,  BATCH * WD * ATT_H)                 // 48000
  WS_ALLOC(c16,   float,  BATCH * ATT_H)                      // 160
  WS_ALLOC(Gsoft, float,  T_DIM * BATCH)                      // 20480
  WS_ALLOC(gi,    float,  (size_t)T_DIM * BATCH * NH)         // 18.43M
  WS_ALLOC(A2,    __bf16, (size_t)BATCH * TT2 * KC2 * 512)    // 19.0M
  WS_ALLOC(B2,    __bf16, (size_t)BATCH * KC2 * 512)          // 148K
  WS_ALLOC(A4,    __bf16, (size_t)(T_DIM * BATCH / 16) * KC45 * 512) // 6.55M
  WS_ALLOC(B4,    __bf16, (size_t)NT45 * KC45 * 512)          // 292K
  WS_ALLOC(B5,    __bf16, (size_t)NT45 * KC45 * 512)          // 292K
#undef WS_ALLOC

  const int TOT_A2 = BATCH * TT2 * KC2 * 512;
  const int TOT_B2 = BATCH * KC2 * 512;
  const int TOT_A4 = (T_DIM * BATCH / 16) * KC45 * 512;
  const int TOT_W  = NT45 * KC45 * 512;

  hipLaunchKernelGGL(k1_precompute, dim3(1), dim3(256), 0, stream,
                     m, q, Wb, W1, b1, Weff, c16);
  hipLaunchKernelGGL(k_pack_gateB, dim3((TOT_B2 + 255) / 256), dim3(256), 0, stream,
                     Weff, W1, B2);
  hipLaunchKernelGGL(k_pack_gateA, dim3((TOT_A2 + 255) / 256), dim3(256), 0, stream,
                     cs, m, q, A2);
  hipLaunchKernelGGL(k_pack_w, dim3((TOT_W + 255) / 256), dim3(256), 0, stream, Wih, B4);
  hipLaunchKernelGGL(k_pack_w, dim3((TOT_W + 255) / 256), dim3(256), 0, stream, Whh, B5);
  hipLaunchKernelGGL(k2_gate, dim3(TT2, BATCH), dim3(32), 0, stream,
                     A2, B2, W2, b2, c16, msk, G);
  hipLaunchKernelGGL(k3_softmax, dim3(BATCH), dim3(256), 0, stream, G, Gsoft);
  hipLaunchKernelGGL(k_pack_giA, dim3((TOT_A4 + 255) / 256), dim3(256), 0, stream,
                     cs, Gsoft, A4);
  hipLaunchKernelGGL(k4_gi, dim3(T_DIM * BATCH / 16, NT45), dim3(32), 0, stream,
                     A4, B4, bih, gi);
  hipLaunchKernelGGL(k5_gru, dim3(1), dim3(1024), 0, stream,
                     m, B5, bhh, gi, hfin);
}
